// Encoder_TRANSFORMER_13469017441009
// MI455X (gfx1250) — matrix operand order repaired and checked by reading, not compiled, not run on hardware
//
#include <hip/hip_runtime.h>
#include <hip/hip_bf16.h>
#include <math.h>

typedef __attribute__((ext_vector_type(16))) _Float16 v16h;
typedef __attribute__((ext_vector_type(8)))  _Float16 v8h;
typedef __attribute__((ext_vector_type(8)))  float    v8f;

#define D_MODEL   32
#define N_HEADS   4
#define HEAD_DIM  8
#define N_LAYERS  4
#define FFDIM     1024
#define BB        2
#define SS        50
#define NN        2500
#define EE        7500
#define GG        (BB*SS)      // 100 graphs
#define TT        (SS+2)       // 52 tokens
#define M_PAD     2512         // 157 tiles of 16
#define TILES_M   (M_PAD/16)
#define MROWS     (TT*BB)      // 104
#define MROWS_PAD 112          // 7 tiles of 16

// ---------------- WMMA fragment builders (CDNA5 ISA 7.12.2) ------------
// A 16x32 f16, lane = M (0..15) + khalf: halves are two contiguous 8-runs:
//   af[0..7]  = Arow[kc + khalf*8      .. +7]
//   af[8..15] = Arow[kc + 16 + khalf*8 .. +7]
__device__ __forceinline__ v16h loadA_frag(const _Float16* Ar, int kc, int khalf) {
  v8h a0 = *(const v8h*)(Ar + kc + khalf * 8);
  v8h a1 = *(const v8h*)(Ar + kc + 16 + khalf * 8);
  return __builtin_shufflevector(a0, a1, 0,1,2,3,4,5,6,7,8,9,10,11,12,13,14,15);
}
// B 32x16 f16, lane = N: same K order as A; with B stored [N][K] (f16, prepped)
// the fragment is two 8-runs at kc + khalf*8 and kc + 16 + khalf*8.
__device__ __forceinline__ v16h loadB_frag(const _Float16* Br, int kc, int khalf) {
  v8h b0 = *(const v8h*)(Br + kc + khalf * 8);
  v8h b1 = *(const v8h*)(Br + kc + 16 + khalf * 8);
  return __builtin_shufflevector(b0, b1, 0,1,2,3,4,5,6,7,8,9,10,11,12,13,14,15);
}

// Monotone-uint encoding for float atomicMax (handles negatives).
__device__ __forceinline__ unsigned encf(float f) {
  unsigned u = __float_as_uint(f);
  return (u & 0x80000000u) ? ~u : (u | 0x80000000u);
}
__device__ __forceinline__ float decf(unsigned u) {
  return (u & 0x80000000u) ? __uint_as_float(u & 0x7fffffffu) : __uint_as_float(~u);
}

// ---------------- generic fills / weight prep ----------------
__global__ void k_zerof(float* p, long n) {
  long t = (long)blockIdx.x * blockDim.x + threadIdx.x;
  if (t < n) p[t] = 0.0f;
}
__global__ void k_zeroh(_Float16* p, long n) {
  long t = (long)blockIdx.x * blockDim.x + threadIdx.x;
  if (t < n) p[t] = (_Float16)0.0f;
}
__global__ void k_fillf(float* p, long n, float v) {
  long t = (long)blockIdx.x * blockDim.x + threadIdx.x;
  if (t < n) p[t] = v;
}
__global__ void k_pool_init(unsigned* p, long n) {
  long t = (long)blockIdx.x * blockDim.x + threadIdx.x;
  if (t < n) p[t] = encf(-3.0e38f);
}
// f32 [K][N] -> f16 [N][K] (transpose+convert) for GCN weights
__global__ void k_cvt_wT(const float* __restrict__ src, _Float16* __restrict__ dst,
                         int K, int N) {
  long t = (long)blockIdx.x * blockDim.x + threadIdx.x;
  if (t >= (long)K * N) return;
  int n = (int)(t % N), k = (int)(t / N);
  dst[(size_t)n * K + k] = (_Float16)src[(size_t)k * N + n];
}
// straight f32 -> f16 convert (layout already [N][K]) for w1/w2
__global__ void k_cvt_w(const float* __restrict__ src, _Float16* __restrict__ dst, long n) {
  long t = (long)blockIdx.x * blockDim.x + threadIdx.x;
  if (t < n) dst[t] = (_Float16)src[t];
}

// ---------------- pointwise embed: x = relu(v @ W_emb + b), f16, pad rows zero ----
__global__ void k_embed(const float* __restrict__ v, const float* __restrict__ W,
                        const float* __restrict__ b, _Float16* __restrict__ x,
                        int g0, int ng) {
  long t = (long)blockIdx.x * blockDim.x + threadIdx.x;
  if (t >= (long)ng * M_PAD) return;
  int m  = (int)(t % M_PAD);
  int gc = (int)(t / M_PAD);
  _Float16* xo = x + ((size_t)gc * M_PAD + m) * 64;
  if (m >= NN) { for (int d = 0; d < 64; ++d) xo[d] = (_Float16)0.0f; return; }
  const float* vp = v + ((size_t)(g0 + gc) * NN + m) * 3;
  float c0 = vp[0], c1 = vp[1], c2 = vp[2];
  for (int d = 0; d < 64; ++d) {
    float a = b[d] + c0 * W[d] + c1 * W[64 + d] + c2 * W[128 + d];
    xo[d] = (_Float16)fmaxf(a, 0.0f);
  }
}

// ---------------- degree (self-loop init 1.0 + per-edge add at col) -------------
__global__ void k_deg_acc(const int* __restrict__ edges, float* __restrict__ deg,
                          int g0, int ng) {
  long t = (long)blockIdx.x * blockDim.x + threadIdx.x;
  if (t >= (long)ng * EE) return;
  int e  = (int)(t % EE);
  int gc = (int)(t / EE);
  int col = edges[((size_t)(g0 + gc) * EE + e) * 2 + 1];
  if ((unsigned)col < NN) atomicAdd(&deg[(size_t)gc * NN + col], 1.0f);
}

// ------- batched WMMA GEMM: C[g] = A[g](f16,[M_PAD,K]) @ Bh^T (Bh f16 [Nc][K]) -----
__global__ void k_gemm_xw(const _Float16* __restrict__ A, const _Float16* __restrict__ Bh,
                          float* __restrict__ C, int K, int Nc, int ng) {
  int tilesN = Nc >> 4;
  int wave = (int)(((long)blockIdx.x * blockDim.x + threadIdx.x) >> 5);
  int lane = threadIdx.x & 31;
  long total = (long)ng * TILES_M * tilesN;
  if (wave >= total) return;                // wave-uniform guard (EXEC all-1 at WMMA)
  int g  = wave / (TILES_M * tilesN);
  int r0 = wave % (TILES_M * tilesN);
  int mt = r0 / tilesN, nt = r0 % tilesN;
  int l15 = lane & 15, khalf = lane >> 4;
  const _Float16* Ar = A + ((size_t)g * M_PAD + (mt * 16 + l15)) * K;
  int ncol = nt * 16 + l15;
  const _Float16* Br = Bh + (size_t)ncol * K;
  v8f acc = {};
  for (int kc = 0; kc < K; kc += 32) {
    if (kc + 32 < K) {
      __builtin_prefetch((const void*)(Ar + kc + 32), 0, 0);
      __builtin_prefetch((const void*)(Br + kc + 32), 0, 0);
    }
    v16h af = loadA_frag(Ar, kc, khalf);
    v16h bf = loadB_frag(Br, kc, khalf);
    acc = __builtin_amdgcn_wmma_f32_16x16x32_f16(false, af, false, bf,
                                                 (short)0, acc, false, false);
  }
  float* Cg = C + (size_t)g * M_PAD * Nc;
#pragma unroll
  for (int r = 0; r < 8; ++r)
    Cg[(size_t)(mt * 16 + r + khalf * 8) * Nc + ncol] = acc[r];
}

// ---------------- edge scatter: agg[col] += dinv[row]*dinv[col]*xw[row] -----------
__global__ void k_scatter(const float* __restrict__ xw, float* __restrict__ agg,
                          const int* __restrict__ edges, const float* __restrict__ deg,
                          int F, int g0, int ng) {
  long t = (long)blockIdx.x * blockDim.x + threadIdx.x;
  if (t >= (long)ng * EE * F) return;
  int f  = (int)(t % F);
  int e  = (int)((t / F) % EE);
  int gc = (int)(t / ((long)F * EE));
  const int* ep = edges + ((size_t)(g0 + gc) * EE + e) * 2;
  int row = ep[0], col = ep[1];
  if ((unsigned)row >= NN || (unsigned)col >= NN) return;
  float w = rsqrtf(deg[(size_t)gc * NN + row]) * rsqrtf(deg[(size_t)gc * NN + col]);
  atomicAdd(&agg[((size_t)gc * M_PAD + col) * F + f],
            w * xw[((size_t)gc * M_PAD + row) * F + f]);
}

// ---------------- combine layer1: h = leaky(agg + dinv^2*xw + b) -> f16 -----------
__global__ void k_combine1(const float* __restrict__ xw, const float* __restrict__ agg,
                           const float* __restrict__ deg, const float* __restrict__ bias,
                           _Float16* __restrict__ h, int F, int ng) {
  long t = (long)blockIdx.x * blockDim.x + threadIdx.x;
  if (t >= (long)ng * M_PAD * F) return;
  int f  = (int)(t % F);
  int m  = (int)((t / F) % M_PAD);
  int gc = (int)(t / ((long)F * M_PAD));
  size_t idx = ((size_t)gc * M_PAD + m) * F + f;
  if (m >= NN) { h[idx] = (_Float16)0.0f; return; }
  float dv = rsqrtf(deg[(size_t)gc * NN + m]);
  float val = agg[idx] + dv * dv * xw[idx] + bias[f];
  val = (val >= 0.0f) ? val : 0.15f * val;
  h[idx] = (_Float16)val;
}

// ---------------- combine layer2 + max-pool over nodes ---------------------------
__global__ void k_combine2(const float* __restrict__ xw, const float* __restrict__ agg,
                           const float* __restrict__ deg, const float* __restrict__ bias,
                           unsigned* __restrict__ pooled, int g0, int ng) {
  long t = (long)blockIdx.x * blockDim.x + threadIdx.x;
  if (t >= (long)ng * NN * 256) return;
  int f  = (int)(t % 256);
  int m  = (int)((t / 256) % NN);
  int gc = (int)(t / ((long)256 * NN));
  size_t idx = ((size_t)gc * M_PAD + m) * 256 + f;
  float dv = rsqrtf(deg[(size_t)gc * NN + m]);
  float val = agg[idx] + dv * dv * xw[idx] + bias[f];
  val = (val >= 0.0f) ? val : 0.15f * val;
  atomicMax(&pooled[(size_t)(g0 + gc) * 256 + f], encf(val));
}

// ---------------- build sequence: [muQ, sigmaQ, relu(pool@W_fc+b)] + pe ----------
__global__ void k_seq(const unsigned* __restrict__ pooled, const float* __restrict__ Wfc,
                      const float* __restrict__ bfc, const float* __restrict__ muQ,
                      const float* __restrict__ sigmaQ, const float* __restrict__ pe,
                      float* __restrict__ xseq) {
  int t = blockIdx.x * blockDim.x + threadIdx.x;
  if (t >= TT * BB * D_MODEL) return;
  int d = t % D_MODEL;
  int b = (t / D_MODEL) % BB;
  int tt = t / (D_MODEL * BB);
  float val;
  if (tt == 0)      val = muQ[d];
  else if (tt == 1) val = sigmaQ[d];
  else {
    int g = b * SS + (tt - 2);
    float acc = bfc[d];
    for (int k = 0; k < 256; ++k)
      acc += decf(pooled[(size_t)g * 256 + k]) * Wfc[k * D_MODEL + d];
    val = fmaxf(acc, 0.0f);
  }
  xseq[t] = val + pe[tt * D_MODEL + d];
}

// ---------------- transformer pieces ---------------------------------------------
__global__ void k_qkv(const float* __restrict__ xseq, const float* __restrict__ iw,
                      const float* __restrict__ ib, float* __restrict__ qkv) {
  int t = blockIdx.x * blockDim.x + threadIdx.x;
  if (t >= TT * BB * 96) return;
  int j   = t % 96;
  int row = t / 96;  // t*BB + b
  float acc = ib[j];
  for (int d = 0; d < D_MODEL; ++d)
    acc += xseq[row * D_MODEL + d] * iw[j * D_MODEL + d];
  qkv[t] = acc;
}

__global__ void k_attn(const float* __restrict__ qkv, float* __restrict__ attno) {
  int t = blockIdx.x * blockDim.x + threadIdx.x;
  if (t >= BB * N_HEADS * TT) return;
  int tt = t % TT;
  int h  = (t / TT) % N_HEADS;
  int b  = t / (TT * N_HEADS);
  const float scale = 0.35355339059327373f; // 1/sqrt(8)
  float logits[TT];
  const float* q = qkv + (tt * BB + b) * 96 + h * HEAD_DIM;
  float mx = -3.0e38f;
  for (int s = 0; s < TT; ++s) {
    const float* k = qkv + (s * BB + b) * 96 + D_MODEL + h * HEAD_DIM;
    float acc = 0.0f;
    for (int d = 0; d < HEAD_DIM; ++d) acc += q[d] * k[d];
    acc *= scale;
    logits[s] = acc;
    mx = fmaxf(mx, acc);
  }
  float sum = 0.0f;
  for (int s = 0; s < TT; ++s) { logits[s] = __expf(logits[s] - mx); sum += logits[s]; }
  float inv = 1.0f / sum;
  for (int d = 0; d < HEAD_DIM; ++d) {
    float acc = 0.0f;
    for (int s = 0; s < TT; ++s)
      acc += logits[s] * qkv[(s * BB + b) * 96 + 2 * D_MODEL + h * HEAD_DIM + d];
    attno[(tt * BB + b) * D_MODEL + h * HEAD_DIM + d] = acc * inv;
  }
}

__global__ void k_ares_ln(const float* __restrict__ attno, const float* __restrict__ ow,
                          const float* __restrict__ ob, const float* __restrict__ g,
                          const float* __restrict__ be, float* __restrict__ xseq,
                          _Float16* __restrict__ xh) {
  int row = blockIdx.x * blockDim.x + threadIdx.x;
  if (row >= MROWS) return;
  float r[D_MODEL];
  for (int d = 0; d < D_MODEL; ++d) {
    float o = ob[d];
    for (int e = 0; e < D_MODEL; ++e)
      o += attno[row * D_MODEL + e] * ow[d * D_MODEL + e];
    r[d] = xseq[row * D_MODEL + d] + o;
  }
  float m = 0.0f;
  for (int d = 0; d < D_MODEL; ++d) m += r[d];
  m *= (1.0f / D_MODEL);
  float var = 0.0f;
  for (int d = 0; d < D_MODEL; ++d) { float dd = r[d] - m; var += dd * dd; }
  var *= (1.0f / D_MODEL);
  float rs = rsqrtf(var + 1e-5f);
  for (int d = 0; d < D_MODEL; ++d) {
    float y = (r[d] - m) * rs * g[d] + be[d];
    xseq[row * D_MODEL + d] = y;
    xh[row * D_MODEL + d] = (_Float16)y;
  }
}

// FF1: [112,32] @ w1^T -> gelu -> f16 [112,1024]; single K=32 WMMA step per tile
__global__ void k_ff1(const _Float16* __restrict__ A, const _Float16* __restrict__ w1h,
                      const float* __restrict__ b1b, _Float16* __restrict__ ff1h) {
  const int tilesN = FFDIM / 16;
  int wave = (int)(((long)blockIdx.x * blockDim.x + threadIdx.x) >> 5);
  int lane = threadIdx.x & 31;
  if (wave >= (MROWS_PAD / 16) * tilesN) return;
  int mt = wave / tilesN, nt = wave % tilesN;
  int l15 = lane & 15, khalf = lane >> 4;
  const _Float16* Ar = A + (mt * 16 + l15) * D_MODEL;
  int ncol = nt * 16 + l15;
  v16h af = loadA_frag(Ar, 0, khalf);
  v16h bf = loadB_frag(w1h + (size_t)ncol * D_MODEL, 0, khalf);
  v8f acc = {};
  acc = __builtin_amdgcn_wmma_f32_16x16x32_f16(false, af, false, bf,
                                               (short)0, acc, false, false);
  float bias = b1b[ncol];
#pragma unroll
  for (int r = 0; r < 8; ++r) {
    int row = mt * 16 + r + khalf * 8;
    float x = acc[r] + bias;
    float gel = 0.5f * x * (1.0f + erff(x * 0.7071067811865476f));
    ff1h[(size_t)row * FFDIM + ncol] = (_Float16)gel;
  }
}

// FF2: [112,1024] @ w2^T + b2 -> f32 [112,32]; 32 WMMA K-steps per tile
__global__ void k_ff2(const _Float16* __restrict__ A, const _Float16* __restrict__ w2h,
                      const float* __restrict__ b2b, float* __restrict__ ff2) {
  const int tilesN = D_MODEL / 16;
  int wave = (int)(((long)blockIdx.x * blockDim.x + threadIdx.x) >> 5);
  int lane = threadIdx.x & 31;
  if (wave >= (MROWS_PAD / 16) * tilesN) return;
  int mt = wave / tilesN, nt = wave % tilesN;
  int l15 = lane & 15, khalf = lane >> 4;
  const _Float16* Ar = A + (size_t)(mt * 16 + l15) * FFDIM;
  int ncol = nt * 16 + l15;
  const _Float16* Br = w2h + (size_t)ncol * FFDIM;
  v8f acc = {};
  for (int kc = 0; kc < FFDIM; kc += 32) {
    if (kc + 32 < FFDIM) {
      __builtin_prefetch((const void*)(Ar + kc + 32), 0, 0);
      __builtin_prefetch((const void*)(Br + kc + 32), 0, 0);
    }
    v16h af = loadA_frag(Ar, kc, khalf);
    v16h bf = loadB_frag(Br, kc, khalf);
    acc = __builtin_amdgcn_wmma_f32_16x16x32_f16(false, af, false, bf,
                                                 (short)0, acc, false, false);
  }
  float bias = b2b[ncol];
#pragma unroll
  for (int r = 0; r < 8; ++r)
    ff2[(size_t)(mt * 16 + r + khalf * 8) * D_MODEL + ncol] = acc[r] + bias;
}

__global__ void k_fres_ln(const float* __restrict__ ff2, const float* __restrict__ g,
                          const float* __restrict__ be, float* __restrict__ xseq,
                          _Float16* __restrict__ xh) {
  int row = blockIdx.x * blockDim.x + threadIdx.x;
  if (row >= MROWS) return;
  float r[D_MODEL];
  for (int d = 0; d < D_MODEL; ++d)
    r[d] = xseq[row * D_MODEL + d] + ff2[row * D_MODEL + d];
  float m = 0.0f;
  for (int d = 0; d < D_MODEL; ++d) m += r[d];
  m *= (1.0f / D_MODEL);
  float var = 0.0f;
  for (int d = 0; d < D_MODEL; ++d) { float dd = r[d] - m; var += dd * dd; }
  var *= (1.0f / D_MODEL);
  float rs = rsqrtf(var + 1e-5f);
  for (int d = 0; d < D_MODEL; ++d) {
    float y = (r[d] - m) * rs * g[d] + be[d];
    xseq[row * D_MODEL + d] = y;
    xh[row * D_MODEL + d] = (_Float16)y;
  }
}

__global__ void k_out(const float* __restrict__ xseq, float* __restrict__ out) {
  int t = blockIdx.x * blockDim.x + threadIdx.x;
  if (t >= 128 + TT * BB * D_MODEL) return;
  out[t] = (t < 128) ? xseq[t] : xseq[t - 128];
}

// ============================ host launch ==========================================
static inline int nblk(long n, int bs) { return (int)((n + bs - 1) / bs); }

extern "C" void kernel_launch(void* const* d_in, const int* in_sizes, int n_in,
                              void* d_out, int out_size, void* d_ws, size_t ws_size,
                              hipStream_t stream) {
  (void)in_sizes; (void)n_in; (void)out_size;
  const float* v      = (const float*)d_in[0];
  const int*   edges  = (const int*)d_in[2];      // d_in[1] = f, unused by reference
  const float* W_emb  = (const float*)d_in[3];
  const float* b_emb  = (const float*)d_in[4];
  const float* W_g1   = (const float*)d_in[5];
  const float* b_g1   = (const float*)d_in[6];
  const float* W_g2   = (const float*)d_in[7];
  const float* b_g2   = (const float*)d_in[8];
  const float* W_fc   = (const float*)d_in[9];
  const float* b_fc   = (const float*)d_in[10];
  const float* muQ    = (const float*)d_in[11];
  const float* sigmaQ = (const float*)d_in[12];
  const float* pe     = (const float*)d_in[13];
  const float* in_w   = (const float*)d_in[14];
  const float* in_b   = (const float*)d_in[15];
  const float* out_w  = (const float*)d_in[16];
  const float* out_b  = (const float*)d_in[17];
  const float* ln1_g  = (const float*)d_in[18];
  const float* ln1_b  = (const float*)d_in[19];
  const float* w1     = (const float*)d_in[20];
  const float* b1     = (const float*)d_in[21];
  const float* w2     = (const float*)d_in[22];
  const float* b2     = (const float*)d_in[23];
  const float* ln2_g  = (const float*)d_in[24];
  const float* ln2_b  = (const float*)d_in[25];
  float* out = (float*)d_out;

  // ---- workspace carve-up (deterministic: depends only on ws_size) ----
  size_t off = 0;
  auto alloc = [&](size_t bytes) -> void* {
    void* p = (char*)d_ws + off;
    off += (bytes + 255) & ~(size_t)255;
    return p;
  };
  unsigned*  pooled = (unsigned*)alloc((size_t)GG * 256 * 4);
  float*     xseq   = (float*)alloc((size_t)TT * BB * D_MODEL * 4);
  _Float16*  xseqh  = (_Float16*)alloc((size_t)MROWS_PAD * D_MODEL * 2);
  float*     qkv    = (float*)alloc((size_t)TT * BB * 96 * 4);
  float*     attno  = (float*)alloc((size_t)TT * BB * D_MODEL * 4);
  _Float16*  ff1h   = (_Float16*)alloc((size_t)MROWS_PAD * FFDIM * 2);
  float*     ff2buf = (float*)alloc((size_t)MROWS_PAD * D_MODEL * 4);
  // pre-converted f16 weights, laid out [N][K] for contiguous B fragments
  _Float16*  g1h    = (_Float16*)alloc((size_t)128 * 64 * 2);
  _Float16*  g2h    = (_Float16*)alloc((size_t)256 * 128 * 2);
  _Float16*  w1h    = (_Float16*)alloc((size_t)N_LAYERS * FFDIM * D_MODEL * 2);
  _Float16*  w2h    = (_Float16*)alloc((size_t)N_LAYERS * D_MODEL * FFDIM * 2);

  // per-graph chunk buffers
  const size_t szX    = (size_t)M_PAD * 64 * 2;
  const size_t szDeg  = (size_t)NN * 4;
  const size_t szXw1  = (size_t)M_PAD * 128 * 4;
  const size_t szH1   = (size_t)M_PAD * 128 * 2;
  const size_t szXw2  = (size_t)M_PAD * 256 * 4;
  const size_t perG   = szX + szDeg + 2 * szXw1 + szH1 + 2 * szXw2 + 8 * 256;
  size_t remain = (ws_size > off) ? (ws_size - off) : 0;
  long Cmax = (long)(remain / perG);
  if (Cmax < 1) Cmax = 1;
  if (Cmax > GG) Cmax = GG;
  int C = (int)Cmax;

  _Float16* xh   = (_Float16*)alloc((size_t)C * szX);
  float*    deg  = (float*)alloc((size_t)C * szDeg);
  float*    xw1  = (float*)alloc((size_t)C * szXw1);
  float*    agg1 = (float*)alloc((size_t)C * szXw1);
  _Float16* h1   = (_Float16*)alloc((size_t)C * szH1);
  float*    xw2  = (float*)alloc((size_t)C * szXw2);
  float*    agg2 = (float*)alloc((size_t)C * szXw2);

  const int BS = 256;

  // one-time weight prep (f32 -> f16, [N][K] layout)
  k_cvt_wT<<<nblk(64L * 128, BS), BS, 0, stream>>>(W_g1, g1h, 64, 128);
  k_cvt_wT<<<nblk(128L * 256, BS), BS, 0, stream>>>(W_g2, g2h, 128, 256);
  k_cvt_w<<<nblk((long)N_LAYERS * FFDIM * D_MODEL, BS), BS, 0, stream>>>(
      w1, w1h, (long)N_LAYERS * FFDIM * D_MODEL);
  k_cvt_w<<<nblk((long)N_LAYERS * D_MODEL * FFDIM, BS), BS, 0, stream>>>(
      w2, w2h, (long)N_LAYERS * D_MODEL * FFDIM);

  k_pool_init<<<nblk((long)GG * 256, BS), BS, 0, stream>>>(pooled, (long)GG * 256);

  for (int g0 = 0; g0 < GG; g0 += C) {
    int ng = (GG - g0 < C) ? (GG - g0) : C;
    // embed
    k_embed<<<nblk((long)ng * M_PAD, BS), BS, 0, stream>>>(v, W_emb, b_emb, xh, g0, ng);
    // degree (self-loop = 1.0 init)
    k_fillf<<<nblk((long)ng * NN, BS), BS, 0, stream>>>(deg, (long)ng * NN, 1.0f);
    k_deg_acc<<<nblk((long)ng * EE, BS), BS, 0, stream>>>(edges, deg, g0, ng);
    // GCN layer 1: xw1 = x @ W_g1  (K=64, N=128), scatter, combine
    k_gemm_xw<<<nblk((long)ng * TILES_M * 8 * 32, BS), BS, 0, stream>>>(xh, g1h, xw1, 64, 128, ng);
    k_zerof<<<nblk((long)ng * M_PAD * 128, BS), BS, 0, stream>>>(agg1, (long)ng * M_PAD * 128);
    k_scatter<<<nblk((long)ng * EE * 128, BS), BS, 0, stream>>>(xw1, agg1, edges, deg, 128, g0, ng);
    k_combine1<<<nblk((long)ng * M_PAD * 128, BS), BS, 0, stream>>>(xw1, agg1, deg, b_g1, h1, 128, ng);
    // GCN layer 2: xw2 = h1 @ W_g2 (K=128, N=256), scatter, combine + max-pool
    k_gemm_xw<<<nblk((long)ng * TILES_M * 16 * 32, BS), BS, 0, stream>>>(h1, g2h, xw2, 128, 256, ng);
    k_zerof<<<nblk((long)ng * M_PAD * 256, BS), BS, 0, stream>>>(agg2, (long)ng * M_PAD * 256);
    k_scatter<<<nblk((long)ng * EE * 256, BS), BS, 0, stream>>>(xw2, agg2, edges, deg, 256, g0, ng);
    k_combine2<<<nblk((long)ng * NN * 256, BS), BS, 0, stream>>>(xw2, agg2, deg, b_g2, pooled, g0, ng);
  }

  // sequence build + transformer
  k_seq<<<nblk(TT * BB * D_MODEL, BS), BS, 0, stream>>>(pooled, W_fc, b_fc, muQ, sigmaQ, pe, xseq);
  k_zeroh<<<nblk((long)MROWS_PAD * D_MODEL, BS), BS, 0, stream>>>(xseqh, (long)MROWS_PAD * D_MODEL);

  for (int i = 0; i < N_LAYERS; ++i) {
    k_qkv<<<nblk(TT * BB * 96, BS), BS, 0, stream>>>(xseq, in_w + (size_t)i * 96 * D_MODEL,
                                                     in_b + i * 96, qkv);
    k_attn<<<nblk(BB * N_HEADS * TT, BS), BS, 0, stream>>>(qkv, attno);
    k_ares_ln<<<1, 128, 0, stream>>>(attno, out_w + (size_t)i * D_MODEL * D_MODEL,
                                     out_b + i * D_MODEL, ln1_g + i * D_MODEL,
                                     ln1_b + i * D_MODEL, xseq, xseqh);
    k_ff1<<<nblk((long)(MROWS_PAD / 16) * (FFDIM / 16) * 32, BS), BS, 0, stream>>>(
        xseqh, w1h + (size_t)i * FFDIM * D_MODEL, b1 + i * FFDIM, ff1h);
    k_ff2<<<nblk((long)(MROWS_PAD / 16) * (D_MODEL / 16) * 32, BS), BS, 0, stream>>>(
        ff1h, w2h + (size_t)i * D_MODEL * FFDIM, b2 + i * D_MODEL, ff2buf);
    k_fres_ln<<<1, 128, 0, stream>>>(ff2buf, ln2_g + i * D_MODEL, ln2_b + i * D_MODEL,
                                     xseq, xseqh);
  }

  k_out<<<nblk(128 + TT * BB * D_MODEL, BS), BS, 0, stream>>>(xseq, out);
}
